// AttentionBlock_37666863186388
// MI455X (gfx1250) — matrix operand order repaired and checked by reading, not compiled, not run on hardware
//
#include <hip/hip_runtime.h>
#include <hip/hip_bf16.h>

// ---------------------------------------------------------------------------
// AttentionBlock for MI455X (gfx1250): bf16 WMMA (16x16x32) everywhere,
// f32 accumulation, flash-attention streaming with async double-buffered
// K/V staging into LDS (ASYNCcnt path). Shapes: B=2, C=256, L=4096,
// HEADS=4, CH=64, GROUPS=32.
// ---------------------------------------------------------------------------

typedef __attribute__((ext_vector_type(16))) __bf16 v16bf;
typedef __attribute__((ext_vector_type(8)))  float  v8f;
typedef int v4i __attribute__((vector_size(16)));

union Frag {
  v16bf v;
  unsigned short u[16];
  uint4 q[2];
};

__device__ __forceinline__ unsigned short f2bf(float f) {
  unsigned int u = __float_as_uint(f);
  u += 0x7FFFu + ((u >> 16) & 1u);           // round-to-nearest-even
  return (unsigned short)(u >> 16);
}

__device__ __forceinline__ v8f wmma_bf16(const Frag& a, const Frag& b, v8f c) {
  // 8 args: (neg_a, A, neg_b, B, c_mod, C, reuse_a, reuse_b)
  return __builtin_amdgcn_wmma_f32_16x16x32_bf16(false, a.v, false, b.v,
                                                 (short)0, c, false, false);
}

// --- async global->LDS copy of 16 bytes per lane (CDNA5 ASYNCcnt path) -----
#if __has_builtin(__builtin_amdgcn_global_load_async_to_lds_b128)
#define HAVE_ASYNC_LDS 1
#else
#define HAVE_ASYNC_LDS 0
#endif

typedef __attribute__((address_space(1))) v4i* gv4i_p;  // global int4*
typedef __attribute__((address_space(3))) v4i* lv4i_p;  // LDS int4*

__device__ __forceinline__ void async_cp16(const unsigned short* g,
                                           unsigned short* l) {
#if HAVE_ASYNC_LDS
  __builtin_amdgcn_global_load_async_to_lds_b128(
      (gv4i_p)(unsigned short*)g, (lv4i_p)l, 0, 0);
#else
  *(uint4*)l = *(const uint4*)g;
#endif
}

__device__ __forceinline__ void wait_async_prev16() {
#if HAVE_ASYNC_LDS
#if __has_builtin(__builtin_amdgcn_s_wait_asynccnt)
  __builtin_amdgcn_s_wait_asynccnt(16);      // 16 newest (next chunk) may fly
#else
  asm volatile("s_wait_asynccnt 0x10" ::: "memory");
#endif
#else
  __syncthreads();                            // single wave: fence only
#endif
}

// ---------------------------------------------------------------------------
// 1) Cast weight matrices f32 -> bf16 (one-time, tiny)
// ---------------------------------------------------------------------------
__global__ __launch_bounds__(256) void k_cast(const float* __restrict__ qkv_w,
                                              const float* __restrict__ proj_w,
                                              unsigned short* __restrict__ qkv_wb,
                                              unsigned short* __restrict__ proj_wb) {
  int i = blockIdx.x * 256 + threadIdx.x;
  if (i < 768 * 256) qkv_wb[i] = f2bf(qkv_w[i]);
  if (i < 256 * 256) proj_wb[i] = f2bf(proj_w[i]);
}

// ---------------------------------------------------------------------------
// 2) GroupNorm over [B,C,L], 32 groups of 8ch x 4096. One block per (b,g).
//    Writes h transposed: h_t[b][l][c] bf16 (c contiguous -> B-matrix columns).
// ---------------------------------------------------------------------------
__global__ __launch_bounds__(256) void k_gnorm(const float* __restrict__ x,
                                               const float* __restrict__ nw,
                                               const float* __restrict__ nb,
                                               unsigned short* __restrict__ h_t) {
  const int b = blockIdx.x >> 5, g = blockIdx.x & 31;
  const float* xg = x + ((size_t)b * 256 + g * 8) * 4096;
  float s = 0.f, ss = 0.f;
  for (int i = threadIdx.x; i < 8 * 4096; i += 256) {
    float v = xg[i]; s += v; ss += v * v;
  }
  __shared__ float rs[256], rq[256];
  rs[threadIdx.x] = s; rq[threadIdx.x] = ss;
  __syncthreads();
  for (int o = 128; o > 0; o >>= 1) {
    if ((int)threadIdx.x < o) {
      rs[threadIdx.x] += rs[threadIdx.x + o];
      rq[threadIdx.x] += rq[threadIdx.x + o];
    }
    __syncthreads();
  }
  const float mean = rs[0] * (1.f / 32768.f);
  const float var  = rq[0] * (1.f / 32768.f) - mean * mean;
  const float rstd = rsqrtf(var + 1e-5f);
  for (int i = threadIdx.x; i < 8 * 4096; i += 256) {
    const int cl = i >> 12, l = i & 4095, c = g * 8 + cl;
    const float hn = (xg[i] - mean) * rstd * nw[c] + nb[c];
    h_t[((size_t)b * 4096 + l) * 256 + c] = f2bf(hn);
  }
}

// ---------------------------------------------------------------------------
// 3) QKV GEMM: [768x256] x [256 x 4096] per batch. One wave per 16x16 D tile.
//    Epilogue: +bias, fold 1/sqrt(sqrt(64)) into q,k; route to attention
//    layouts: q_t/k_t = [bh][l][64] (c contiguous), v_m = [bh][64][l].
// ---------------------------------------------------------------------------
__global__ __launch_bounds__(32) void k_qkv(const unsigned short* __restrict__ wq,
                                            const unsigned short* __restrict__ h_t,
                                            const float* __restrict__ bias,
                                            unsigned short* __restrict__ q_t,
                                            unsigned short* __restrict__ k_t,
                                            unsigned short* __restrict__ v_m) {
  const int lt = blockIdx.x, ot = blockIdx.y, b = blockIdx.z;
  const int lane = threadIdx.x, half = lane >> 4, lr = lane & 15;
  const unsigned short* arow = wq + (size_t)(ot * 16 + lr) * 256;
  const unsigned short* brow = h_t + ((size_t)b * 4096 + lt * 16 + lr) * 256;
  v8f acc = {};
#pragma unroll
  for (int kk = 0; kk < 8; ++kk) {        // K = 256, 8 steps of 32
    Frag a, bf;
    a.q[0]  = *(const uint4*)(arow + kk * 32 + half * 16);
    a.q[1]  = *(const uint4*)(arow + kk * 32 + half * 16 + 8);
    bf.q[0] = *(const uint4*)(brow + kk * 32 + half * 16);
    bf.q[1] = *(const uint4*)(brow + kk * 32 + half * 16 + 8);
    acc = wmma_bf16(a, bf, acc);
  }
  const int l = lt * 16 + lr;
  const float scale = 0.3535533905932738f;   // 1/sqrt(sqrt(64))
#pragma unroll
  for (int v = 0; v < 8; ++v) {
    const int o = ot * 16 + v + 8 * half;
    float val = acc[v] + bias[o];
    if (o < 512) val *= scale;
    const unsigned short bv = f2bf(val);
    if (o < 256) {
      const int head = o >> 6, c = o & 63, bh = b * 4 + head;
      q_t[((size_t)bh * 4096 + l) * 64 + c] = bv;
    } else if (o < 512) {
      const int oo = o - 256, head = oo >> 6, c = oo & 63, bh = b * 4 + head;
      k_t[((size_t)bh * 4096 + l) * 64 + c] = bv;
    } else {
      const int oo = o - 512, head = oo >> 6, c = oo & 63, bh = b * 4 + head;
      v_m[((size_t)bh * 64 + c) * 4096 + l] = bv;
    }
  }
}

// ---------------------------------------------------------------------------
// 4) Flash attention: one wave per (bh, 16-row t tile). Streams s in chunks
//    of 32 with ASYNC double-buffered K/V staging in LDS:
//      prefetch(chunk n+1) -> s_wait_asynccnt 16 -> QK^T (4 WMMA from LDS)
//      -> online softmax -> P via LDS transpose -> PV (4 WMMA, K=32).
//    LDS/workgroup: 2x4KB K + 2x4KB V + 1KB P = 17KB.
// ---------------------------------------------------------------------------
__global__ __launch_bounds__(32) void k_attn(const unsigned short* __restrict__ q_t,
                                             const unsigned short* __restrict__ k_t,
                                             const unsigned short* __restrict__ v_m,
                                             unsigned short* __restrict__ a_t) {
  __shared__ __align__(16) unsigned short kBuf[2][32 * 64];  // [s 32][c 64]
  __shared__ __align__(16) unsigned short vBuf[2][64 * 32];  // [c 64][s 32]
  __shared__ __align__(16) unsigned short pLds[16 * 32];
  const int tt = blockIdx.x, bh = blockIdx.y;
  const int lane = threadIdx.x, half = lane >> 4, lr = lane & 15;

  // Persistent Q A-fragments: K(c) = 0..31 and 32..63
  Frag qa0, qa1;
  const unsigned short* qrow = q_t + ((size_t)bh * 4096 + tt * 16 + lr) * 64;
  qa0.q[0] = *(const uint4*)(qrow + half * 16);
  qa0.q[1] = *(const uint4*)(qrow + half * 16 + 8);
  qa1.q[0] = *(const uint4*)(qrow + 32 + half * 16);
  qa1.q[1] = *(const uint4*)(qrow + 32 + half * 16 + 8);

  const unsigned short* kbase = k_t + (size_t)bh * 4096 * 64;
  const unsigned short* vbase = v_m + (size_t)bh * 64 * 4096;

  // Prefetch one 32-s chunk of K (32x64) and V (64x32): 16 async b128 ops.
  auto prefetch = [&](int s0, unsigned short* kb, unsigned short* vb) {
    const unsigned short* kg = kbase + (size_t)(s0 + lane) * 64;
    unsigned short* kl = kb + lane * 64;
#pragma unroll
    for (int it = 0; it < 8; ++it) async_cp16(kg + it * 8, kl + it * 8);
#pragma unroll
    for (int it = 0; it < 8; ++it) {
      const int idx = it * 32 + lane, row = idx >> 2, ch = idx & 3;
      async_cp16(vbase + (size_t)row * 4096 + s0 + ch * 8,
                 vb + row * 32 + ch * 8);
    }
  };

  float mrow[8], lrow[8];
  v8f acc[4];
  const v8f vzero = {};
#pragma unroll
  for (int v = 0; v < 8; ++v) { mrow[v] = -1e30f; lrow[v] = 0.f; }
#pragma unroll
  for (int j = 0; j < 4; ++j) acc[j] = vzero;

  prefetch(0, kBuf[0], vBuf[0]);

#pragma unroll 2
  for (int sc = 0; sc < 128; ++sc) {
    const int cur = sc & 1;
    // Issue next chunk (wraps harmlessly on last iteration), then wait for
    // the current chunk's 16 async ops (oldest) to land in LDS.
    prefetch(((sc + 1) & 127) * 32, kBuf[cur ^ 1], vBuf[cur ^ 1]);
    wait_async_prev16();

    // QK^T: two 16-col score tiles from LDS (B cols = s, contiguous K = c)
    v8f w[2];
#pragma unroll
    for (int j = 0; j < 2; ++j) {
      const unsigned short* kr = &kBuf[cur][(j * 16 + lr) * 64];
      Frag kb0, kb1;
      kb0.q[0] = *(const uint4*)(kr + half * 16);
      kb0.q[1] = *(const uint4*)(kr + half * 16 + 8);
      kb1.q[0] = *(const uint4*)(kr + 32 + half * 16);
      kb1.q[1] = *(const uint4*)(kr + 32 + half * 16 + 8);
      v8f t = vzero;
      t = wmma_bf16(qa0, kb0, t);
      t = wmma_bf16(qa1, kb1, t);
      w[j] = t;
    }
    // Online softmax; row t = v + 8*half lives across the 16 lanes of a half.
#pragma unroll
    for (int v = 0; v < 8; ++v) {
      float m = fmaxf(w[0][v], w[1][v]);
#pragma unroll
      for (int off = 1; off < 16; off <<= 1)
        m = fmaxf(m, __shfl_xor(m, off, 16));
      const float mn = fmaxf(mrow[v], m);
      const float r  = __expf(mrow[v] - mn);
      const float p0 = __expf(w[0][v] - mn);
      const float p1 = __expf(w[1][v] - mn);
      float ps = p0 + p1;
#pragma unroll
      for (int off = 1; off < 16; off <<= 1)
        ps += __shfl_xor(ps, off, 16);
      lrow[v] = lrow[v] * r + ps;
      mrow[v] = mn;
#pragma unroll
      for (int j = 0; j < 4; ++j) acc[j][v] *= r;
      pLds[(v + 8 * half) * 32 + lr]      = f2bf(p0);
      pLds[(v + 8 * half) * 32 + 16 + lr] = f2bf(p1);
    }
    __syncthreads();   // single wave: S_NOP + compiler fence for P transpose
    Frag pa;           // P as A-fragment: row = lr, K runs per half
    pa.q[0] = *(const uint4*)(&pLds[lr * 32 + half * 16]);
    pa.q[1] = *(const uint4*)(&pLds[lr * 32 + half * 16 + 8]);
    // PV: 4 c-tiles (K = this 32-s chunk), V B-fragments from LDS
#pragma unroll
    for (int j = 0; j < 4; ++j) {
      const unsigned short* vr = &vBuf[cur][(j * 16 + lr) * 32 + half * 16];
      Frag vb;
      vb.q[0] = *(const uint4*)(vr);
      vb.q[1] = *(const uint4*)(vr + 8);
      acc[j] = wmma_bf16(pa, vb, acc[j]);
    }
    __syncthreads();
  }
  // Epilogue: normalize by row sum, store a_t[b][l][256] (c contiguous)
  const int b = bh >> 2, head = bh & 3;
#pragma unroll
  for (int v = 0; v < 8; ++v) {
    const float inv = 1.f / lrow[v];
    const int tg = tt * 16 + v + 8 * half;
#pragma unroll
    for (int j = 0; j < 4; ++j) {
      const int c = head * 64 + j * 16 + lr;
      a_t[((size_t)b * 4096 + tg) * 256 + c] = f2bf(acc[j][v] * inv);
    }
  }
}

// ---------------------------------------------------------------------------
// 5) Output projection GEMM [256x256] x [256x4096] + bias + residual (f32 out)
// ---------------------------------------------------------------------------
__global__ __launch_bounds__(32) void k_proj(const unsigned short* __restrict__ wp,
                                             const unsigned short* __restrict__ a_t,
                                             const float* __restrict__ bias,
                                             const float* __restrict__ x,
                                             float* __restrict__ out) {
  const int lt = blockIdx.x, ot = blockIdx.y, b = blockIdx.z;
  const int lane = threadIdx.x, half = lane >> 4, lr = lane & 15;
  const unsigned short* arow = wp + (size_t)(ot * 16 + lr) * 256;
  const unsigned short* brow = a_t + ((size_t)b * 4096 + lt * 16 + lr) * 256;
  v8f acc = {};
#pragma unroll
  for (int kk = 0; kk < 8; ++kk) {
    Frag a, bf;
    a.q[0]  = *(const uint4*)(arow + kk * 32 + half * 16);
    a.q[1]  = *(const uint4*)(arow + kk * 32 + half * 16 + 8);
    bf.q[0] = *(const uint4*)(brow + kk * 32 + half * 16);
    bf.q[1] = *(const uint4*)(brow + kk * 32 + half * 16 + 8);
    acc = wmma_bf16(a, bf, acc);
  }
  const int l = lt * 16 + lr;
#pragma unroll
  for (int v = 0; v < 8; ++v) {
    const int o = ot * 16 + v + 8 * half;
    const size_t idx = ((size_t)b * 256 + o) * 4096 + l;
    out[idx] = acc[v] + bias[o] + x[idx];
  }
}

// ---------------------------------------------------------------------------
// Launch
// ---------------------------------------------------------------------------
extern "C" void kernel_launch(void* const* d_in, const int* in_sizes, int n_in,
                              void* d_out, int out_size, void* d_ws, size_t ws_size,
                              hipStream_t stream) {
  (void)in_sizes; (void)n_in; (void)out_size; (void)ws_size;
  const float* x      = (const float*)d_in[0];
  const float* norm_w = (const float*)d_in[1];
  const float* norm_b = (const float*)d_in[2];
  const float* qkv_w  = (const float*)d_in[3];
  const float* qkv_b  = (const float*)d_in[4];
  const float* proj_w = (const float*)d_in[5];
  const float* proj_b = (const float*)d_in[6];
  float* out = (float*)d_out;

  // Workspace layout (bytes): ~20.5 MB total
  char* ws = (char*)d_ws;
  unsigned short* qkv_wb  = (unsigned short*)(ws + 0);          // 768*256*2
  unsigned short* proj_wb = (unsigned short*)(ws + 393216);     // 256*256*2
  unsigned short* h_t     = (unsigned short*)(ws + 524288);     // 2*4096*256*2
  unsigned short* q_t     = (unsigned short*)(ws + 4718592);    // 8*4096*64*2
  unsigned short* k_t     = (unsigned short*)(ws + 8912896);    // 8*4096*64*2
  unsigned short* v_m     = (unsigned short*)(ws + 13107200);   // 8*64*4096*2
  unsigned short* a_t     = (unsigned short*)(ws + 17301504);   // 2*4096*256*2

  k_cast <<<768, 256, 0, stream>>>(qkv_w, proj_w, qkv_wb, proj_wb);
  k_gnorm<<<64, 256, 0, stream>>>(x, norm_w, norm_b, h_t);
  k_qkv  <<<dim3(256, 48, 2), 32, 0, stream>>>(qkv_wb, h_t, qkv_b, q_t, k_t, v_m);
  k_attn <<<dim3(256, 8), 32, 0, stream>>>(q_t, k_t, v_m, a_t);
  k_proj <<<dim3(256, 16, 2), 32, 0, stream>>>(proj_wb, a_t, proj_b, x, out);
}
